// TransitionDown_29480655520249
// MI455X (gfx1250) — matrix operand order repaired and checked by reading, not compiled, not run on hardware
//
#include <hip/hip_runtime.h>
#include <hip/hip_bf16.h>

typedef __attribute__((ext_vector_type(16))) _Float16 v16h;
typedef __attribute__((ext_vector_type(8)))  float    v8f;

union V16U { v16h v; float4 f[2]; };
union PK4U { unsigned long long u; _Float16 h[4]; };

#define B_   16
#define N_   4096
#define S_   1024
#define K_   16
#define C_   64
#define IN_  67
#define INP_ 96
#define OUT_ 128
#define EPS_ 1e-5f
#define WTF_ (3 * 8 * 32 * 16)   // kstep x ntile x lane x elem, fragment order

// ---------------------------------------------------------------------------
// 0) zero BN accumulators + build fragment-ordered f16 weights:
//    WtF[((ks*8+nt)*32+lane)*16+e] = W[col*67+K], K=ks*32+(lane>>4)*16+e,
//    col = nt*16+(lane&15); zero-padded K>=67. Each lane's frag = 32 contig B.
// ---------------------------------------------------------------------------
__global__ __launch_bounds__(256) void init_kernel(const float* __restrict__ W,
                                                   _Float16* __restrict__ WtF,
                                                   float* __restrict__ gsum,
                                                   float* __restrict__ gsq) {
  int t = blockIdx.x * 256 + threadIdx.x;
  if (blockIdx.x == 0 && threadIdx.x < OUT_) { gsum[threadIdx.x] = 0.f; gsq[threadIdx.x] = 0.f; }
  if (t < WTF_) {
    int e    = t & 15;
    int lane = (t >> 4) & 31;
    int nt   = (t >> 9) & 7;
    int ks   = t >> 12;
    int Kk   = ks * 32 + (lane >> 4) * 16 + e;
    int col  = nt * 16 + (lane & 15);
    float v  = (Kk < IN_) ? W[col * IN_ + Kk] : 0.f;
    WtF[t] = (_Float16)v;
  }
}

// ---------------------------------------------------------------------------
// 1) furthest point sampling: one workgroup per batch, LDS dist + tree argmax
// ---------------------------------------------------------------------------
__global__ __launch_bounds__(256) void fps_kernel(const float* __restrict__ xyz,
                                                  int* __restrict__ fps_idx,
                                                  float* __restrict__ new_xyz) {
  __shared__ float dist[N_];
  __shared__ float rv[256];
  __shared__ int   ri[256];
  const int b = blockIdx.x;
  const int tid = threadIdx.x;
  const float* px = xyz + (size_t)b * N_ * 3;

  for (int j = tid; j < N_; j += 256) dist[j] = 1e10f;
  __syncthreads();

  int last = 0;
  for (int it = 0; it < S_; ++it) {
    if (tid == 0) fps_idx[b * S_ + it] = last;
    if (tid < 3)  new_xyz[(b * S_ + it) * 3 + tid] = px[last * 3 + tid];
    const float lx = px[last * 3 + 0], ly = px[last * 3 + 1], lz = px[last * 3 + 2];

    float bm = -1.f; int bidx = 0;
    for (int j = tid; j < N_; j += 256) {
      float dx = px[j * 3 + 0] - lx, dy = px[j * 3 + 1] - ly, dz = px[j * 3 + 2] - lz;
      float d  = dx * dx + dy * dy + dz * dz;
      float nd = fminf(dist[j], d);
      dist[j] = nd;
      if (nd > bm) { bm = nd; bidx = j; }
    }
    rv[tid] = bm; ri[tid] = bidx;
    __syncthreads();
    for (int off = 128; off > 0; off >>= 1) {
      if (tid < off) {
        float ov = rv[tid + off]; int oi = ri[tid + off];
        if (ov > rv[tid] || (ov == rv[tid] && oi < ri[tid])) { rv[tid] = ov; ri[tid] = oi; }
      }
      __syncthreads();
    }
    last = ri[0];
    __syncthreads();
  }
}

// ---------------------------------------------------------------------------
// 2) brute-force 16-NN; full batch cloud cached in LDS (48 KB)
// ---------------------------------------------------------------------------
__global__ __launch_bounds__(256) void knn_kernel(const float* __restrict__ xyz,
                                                  const float* __restrict__ newxyz,
                                                  int* __restrict__ knn) {
  __shared__ float sx[N_ * 3];
  const int b    = blockIdx.x >> 2;
  const int part = blockIdx.x & 3;
  const int tid  = threadIdx.x;

  for (int j = tid; j < N_ * 3; j += 256) sx[j] = xyz[(size_t)b * N_ * 3 + j];
  __syncthreads();

  const int s = part * 256 + tid;
  const int item = b * S_ + s;
  const float qx = newxyz[item * 3 + 0];
  const float qy = newxyz[item * 3 + 1];
  const float qz = newxyz[item * 3 + 2];

  float bd[K_]; int bi[K_];
  float wv = -1.f; int wp = 0;
#pragma unroll
  for (int k = 0; k < K_; ++k) {
    float dx = sx[k * 3 + 0] - qx, dy = sx[k * 3 + 1] - qy, dz = sx[k * 3 + 2] - qz;
    float d = dx * dx + dy * dy + dz * dz;
    bd[k] = d; bi[k] = k;
    if (d > wv) { wv = d; wp = k; }
  }
  for (int j = K_; j < N_; ++j) {
    float dx = sx[j * 3 + 0] - qx, dy = sx[j * 3 + 1] - qy, dz = sx[j * 3 + 2] - qz;
    float d = dx * dx + dy * dy + dz * dz;
    if (d < wv) {
      bd[wp] = d; bi[wp] = j;
      wv = bd[0]; wp = 0;
#pragma unroll
      for (int k = 1; k < K_; ++k) if (bd[k] > wv) { wv = bd[k]; wp = k; }
    }
  }
  for (int k = 0; k < K_; ++k) knn[item * K_ + k] = bi[k];
}

// ---------------------------------------------------------------------------
// shared helpers for the two WMMA passes
// ---------------------------------------------------------------------------
__device__ __forceinline__ void stage_A(_Float16* __restrict__ sArow,  // [K_*INP_]
                                        const int* __restrict__ sIdx,  // [K_]
                                        const float* __restrict__ sQ,  // [3]
                                        const float* __restrict__ xyz,
                                        const float* __restrict__ feat,
                                        int b, int lane) {
  // 16 rows x 96 halves = 384 4-half chunks; 12 per lane, 64-bit DS stores
  for (int m = 0; m < 12; ++m) {
    int c4 = lane + m * 32;
    int k  = c4 / 24;
    int ci = (c4 - k * 24) * 4;
    int g  = sIdx[k];
    PK4U pk;
#pragma unroll
    for (int q = 0; q < 4; ++q) {
      int i = ci + q;
      float v;
      if (i < 3)        v = xyz[((size_t)b * N_ + g) * 3 + i] - sQ[i];
      else if (i < IN_) v = feat[((size_t)b * N_ + g) * C_ + (i - 3)];
      else              v = 0.f;
      pk.h[q] = (_Float16)v;
    }
    *(unsigned long long*)&sArow[k * INP_ + ci] = pk.u;
  }
}

__device__ __forceinline__ void load_A_frags(const _Float16* __restrict__ sArow,
                                             int row, int hi,
                                             v16h& a0, v16h& a1, v16h& a2) {
  // A 16x32 f16 layout (matches WtF): elem e -> K=16*hi+e
  V16U u0, u1, u2;
  const _Float16* base = &sArow[row * INP_];
  u0.f[0] = *(const float4*)(base +      16 * hi);
  u0.f[1] = *(const float4*)(base + 8 + 16 * hi);
  u1.f[0] = *(const float4*)(base + 32 +      16 * hi);
  u1.f[1] = *(const float4*)(base + 32 + 8 + 16 * hi);
  u2.f[0] = *(const float4*)(base + 64 +      16 * hi);
  u2.f[1] = *(const float4*)(base + 64 + 8 + 16 * hi);
  a0 = u0.v; a1 = u1.v; a2 = u2.v;
}

__device__ __forceinline__ v16h load_B_frag(const _Float16* __restrict__ sWtF,
                                            int ks, int nt, int lane) {
  V16U u;
  const _Float16* bp = &sWtF[(((ks * 8) + nt) * 32 + lane) * 16];  // 32 contig bytes
  u.f[0] = ((const float4*)bp)[0];
  u.f[1] = ((const float4*)bp)[1];
  return u.v;
}

// ---------------------------------------------------------------------------
// 3) WMMA GEMM pass A: h = grouped @ W^T, per-channel sum / sumsq
//    one wave == one (b,s); its 16 neighbors are the M=16 rows of the tile
// ---------------------------------------------------------------------------
__global__ __launch_bounds__(256) void gemm_stats_kernel(const float* __restrict__ xyz,
                                                         const float* __restrict__ feat,
                                                         const float* __restrict__ newxyz,
                                                         const int* __restrict__ knn,
                                                         const _Float16* __restrict__ WtF,
                                                         float* __restrict__ gsum,
                                                         float* __restrict__ gsq) {
  __shared__ __align__(16) _Float16 sWtF[WTF_];       // 24 KB
  __shared__ __align__(16) _Float16 sA[8][K_ * INP_]; // 24 KB
  __shared__ int      sIdx[8][K_];
  __shared__ float    sQ[8][3];
  __shared__ float    sSum[OUT_], sSqr[OUT_];

  const int tid  = threadIdx.x;
  const int w    = tid >> 5;
  const int lane = tid & 31;
  const int row  = lane & 15;
  const int hi   = lane >> 4;

  for (int t = tid; t < WTF_ / 8; t += 256)           // float4 copy (16 B each)
    ((float4*)sWtF)[t] = ((const float4*)WtF)[t];
  if (tid < OUT_) { sSum[tid] = 0.f; sSqr[tid] = 0.f; }
  __syncthreads();

  float acc1[8], acc2[8];
#pragma unroll
  for (int nt = 0; nt < 8; ++nt) { acc1[nt] = 0.f; acc2[nt] = 0.f; }

  const int gw = blockIdx.x * 8 + w;                  // 4096 waves total
  for (int it = 0; it < 4; ++it) {
    const int item = gw + it * 4096;                  // 0..16383
    const int b = item >> 10;

    if (lane < K_) sIdx[w][lane] = knn[item * K_ + lane];
    if (lane < 3)  sQ[w][lane]   = newxyz[item * 3 + lane];
    __syncthreads();
    stage_A(sA[w], sIdx[w], sQ[w], xyz, feat, b, lane);
    __syncthreads();

    v16h a0, a1, a2;
    load_A_frags(sA[w], row, hi, a0, a1, a2);

#pragma unroll
    for (int nt = 0; nt < 8; ++nt) {
      v16h b0 = load_B_frag(sWtF, 0, nt, lane);
      v16h b1 = load_B_frag(sWtF, 1, nt, lane);
      v16h b2 = load_B_frag(sWtF, 2, nt, lane);
      v8f c = {};
      c = __builtin_amdgcn_wmma_f32_16x16x32_f16(false, a0, false, b0, (short)0, c, false, false);
      c = __builtin_amdgcn_wmma_f32_16x16x32_f16(false, a1, false, b1, (short)0, c, false, false);
      c = __builtin_amdgcn_wmma_f32_16x16x32_f16(false, a2, false, b2, (short)0, c, false, false);
      float s1 = 0.f, s2 = 0.f;
#pragma unroll
      for (int r = 0; r < 8; ++r) { float v = c[r]; s1 += v; s2 += v * v; }
      acc1[nt] += s1;
      acc2[nt] += s2;
    }
    __syncthreads();
  }

  // fold the two M-halves (same channel column), then one LDS atomic per nt
#pragma unroll
  for (int nt = 0; nt < 8; ++nt) {
    float s1 = acc1[nt] + __shfl_xor(acc1[nt], 16, 32);
    float s2 = acc2[nt] + __shfl_xor(acc2[nt], 16, 32);
    if (lane < 16) {
      atomicAdd(&sSum[nt * 16 + row], s1);
      atomicAdd(&sSqr[nt * 16 + row], s2);
    }
  }
  __syncthreads();
  if (tid < OUT_) { atomicAdd(&gsum[tid], sSum[tid]); atomicAdd(&gsq[tid], sSqr[tid]); }
}

// ---------------------------------------------------------------------------
// 3b) fold sums into BN scale/shift
// ---------------------------------------------------------------------------
__global__ void finalize_kernel(const float* __restrict__ gsum, const float* __restrict__ gsq,
                                const float* __restrict__ gamma, const float* __restrict__ beta,
                                float* __restrict__ scale, float* __restrict__ shift) {
  int t = threadIdx.x;                    // 128 threads
  const float cnt = (float)(B_ * S_ * K_);
  float m   = gsum[t] / cnt;
  float var = gsq[t] / cnt - m * m;
  float inv = rsqrtf(var + EPS_);
  float sc  = inv * gamma[t];
  scale[t]  = sc;
  shift[t]  = beta[t] - m * sc;
}

// ---------------------------------------------------------------------------
// 4) WMMA GEMM pass B: recompute h, affine + ReLU, max over 16 neighbors
// ---------------------------------------------------------------------------
__global__ __launch_bounds__(256) void gemm_out_kernel(const float* __restrict__ xyz,
                                                       const float* __restrict__ feat,
                                                       const float* __restrict__ newxyz,
                                                       const int* __restrict__ knn,
                                                       const _Float16* __restrict__ WtF,
                                                       const float* __restrict__ scale,
                                                       const float* __restrict__ shift,
                                                       float* __restrict__ out_feat) {
  __shared__ __align__(16) _Float16 sWtF[WTF_];
  __shared__ __align__(16) _Float16 sA[8][K_ * INP_];
  __shared__ int      sIdx[8][K_];
  __shared__ float    sQ[8][3];
  __shared__ float    sScale[OUT_], sShift[OUT_];

  const int tid  = threadIdx.x;
  const int w    = tid >> 5;
  const int lane = tid & 31;
  const int row  = lane & 15;
  const int hi   = lane >> 4;

  for (int t = tid; t < WTF_ / 8; t += 256)
    ((float4*)sWtF)[t] = ((const float4*)WtF)[t];
  if (tid < OUT_) { sScale[tid] = scale[tid]; sShift[tid] = shift[tid]; }
  __syncthreads();

  const int gw = blockIdx.x * 8 + w;
  for (int it = 0; it < 4; ++it) {
    const int item = gw + it * 4096;
    const int b = item >> 10;

    if (lane < K_) sIdx[w][lane] = knn[item * K_ + lane];
    if (lane < 3)  sQ[w][lane]   = newxyz[item * 3 + lane];
    __syncthreads();
    stage_A(sA[w], sIdx[w], sQ[w], xyz, feat, b, lane);
    __syncthreads();

    v16h a0, a1, a2;
    load_A_frags(sA[w], row, hi, a0, a1, a2);

#pragma unroll
    for (int nt = 0; nt < 8; ++nt) {
      v16h b0 = load_B_frag(sWtF, 0, nt, lane);
      v16h b1 = load_B_frag(sWtF, 1, nt, lane);
      v16h b2 = load_B_frag(sWtF, 2, nt, lane);
      v8f c = {};
      c = __builtin_amdgcn_wmma_f32_16x16x32_f16(false, a0, false, b0, (short)0, c, false, false);
      c = __builtin_amdgcn_wmma_f32_16x16x32_f16(false, a1, false, b1, (short)0, c, false, false);
      c = __builtin_amdgcn_wmma_f32_16x16x32_f16(false, a2, false, b2, (short)0, c, false, false);

      const int col = nt * 16 + row;
      const float sc = sScale[col], sh = sShift[col];
      float m = -1e30f;
#pragma unroll
      for (int r = 0; r < 8; ++r) {
        float y = fmaxf(c[r] * sc + sh, 0.f);   // BN affine + ReLU
        m = fmaxf(m, y);
      }
      m = fmaxf(m, __shfl_xor(m, 16, 32));       // max over all 16 neighbors
      if (lane < 16) out_feat[(size_t)item * OUT_ + col] = m;
    }
    __syncthreads();
  }
}

// ---------------------------------------------------------------------------
// 5) temb[:, :S, :] strided copy
// ---------------------------------------------------------------------------
__global__ __launch_bounds__(256) void temb_kernel(const float* __restrict__ temb,
                                                   float* __restrict__ out) {
  int t = blockIdx.x * 256 + threadIdx.x;   // B*S*64 = 1048576
  if (t < B_ * S_ * C_) {
    int c = t & 63;
    int s = (t >> 6) & (S_ - 1);
    int b = t >> 16;
    out[t] = temb[((size_t)b * N_ + s) * C_ + c];
  }
}

// ---------------------------------------------------------------------------
extern "C" void kernel_launch(void* const* d_in, const int* in_sizes, int n_in,
                              void* d_out, int out_size, void* d_ws, size_t ws_size,
                              hipStream_t stream) {
  const float* xyz   = (const float*)d_in[0];
  const float* feat  = (const float*)d_in[1];
  const float* temb  = (const float*)d_in[2];
  const float* W     = (const float*)d_in[3];
  const float* gamma = (const float*)d_in[4];
  const float* beta  = (const float*)d_in[5];

  float* out       = (float*)d_out;
  float* o_newxyz  = out;                          // [B,S,3]
  float* o_feat    = out + B_ * S_ * 3;            // [B,S,128]
  float* o_temb    = o_feat + B_ * S_ * OUT_;      // [B,S,64]

  char* ws = (char*)d_ws;
  int*      fpsIdx = (int*)ws;        ws += (size_t)B_ * S_ * 4;
  int*      knnIdx = (int*)ws;        ws += (size_t)B_ * S_ * K_ * 4;
  _Float16* WtF    = (_Float16*)ws;   ws += (size_t)WTF_ * 2;
  float*    gsum   = (float*)ws;      ws += OUT_ * 4;
  float*    gsq    = (float*)ws;      ws += OUT_ * 4;
  float*    scale  = (float*)ws;      ws += OUT_ * 4;
  float*    shift  = (float*)ws;      ws += OUT_ * 4;

  init_kernel<<<48, 256, 0, stream>>>(W, WtF, gsum, gsq);
  fps_kernel<<<B_, 256, 0, stream>>>(xyz, fpsIdx, o_newxyz);
  knn_kernel<<<B_ * 4, 256, 0, stream>>>(xyz, o_newxyz, knnIdx);
  gemm_stats_kernel<<<512, 256, 0, stream>>>(xyz, feat, o_newxyz, knnIdx, WtF, gsum, gsq);
  finalize_kernel<<<1, 128, 0, stream>>>(gsum, gsq, gamma, beta, scale, shift);
  gemm_out_kernel<<<512, 256, 0, stream>>>(xyz, feat, o_newxyz, knnIdx, WtF, scale, shift, o_feat);
  temb_kernel<<<(B_ * S_ * C_ + 255) / 256, 256, 0, stream>>>(temb, o_temb);
}
